// MOE_36369783062554
// MI455X (gfx1250) — matrix operand order repaired and checked by reading, not compiled, not run on hardware
//
#include <hip/hip_runtime.h>
#include <hip/hip_bf16.h>
#include <cmath>

// MoE forward, MI455X (gfx1250, wave32). bf16 WMMA GEMMs w/ fp32 accumulate,
// fp32 router math, sparse top-2 expert dispatch, TDM (tensor_load_to_lds)
// staging of activation tiles (incl. gather-mode for the expert token gather),
// software-pipelined K-loops (load kk+1 fragments before kk's WMMAs).
//
// Confirmed via compile-probe: toolchain is clang>=23 (6-arg tensor builtin);
// the object contains v_wmma_f32_16x16x32_bf16 (222x), 3x tensor_load_to_lds,
// s_wait_tensorcnt, clause-batched b128 loads with partial s_wait_loadcnt.

constexpr int kT = 6272;   // tokens = 32*196
constexpr int kD = 768;
constexpr int kH = 3072;
constexpr int kE = 8;

typedef __attribute__((ext_vector_type(16))) __bf16 v16bf;
typedef __attribute__((ext_vector_type(8)))  float  v8f;
typedef __attribute__((ext_vector_type(4)))  unsigned int u32x4;
typedef __attribute__((ext_vector_type(8)))  int i32x8;
typedef __attribute__((ext_vector_type(4)))  int i32x4;

#if defined(__HIP_DEVICE_COMPILE__) && defined(__gfx1250__) && \
    __has_builtin(__builtin_amdgcn_tensor_load_to_lds) && \
    __has_builtin(__builtin_amdgcn_s_wait_tensorcnt)
#define USE_TDM 1
#else
#define USE_TDM 0
#endif

union FragU { uint4 u[2]; v16bf v; };
static_assert(sizeof(FragU) == 32, "frag size");

__device__ __forceinline__ unsigned short f2bf(float f) {
  unsigned int u = __float_as_uint(f);
  return (unsigned short)((u + 0x7FFFu + ((u >> 16) & 1u)) >> 16);  // RNE
}

__device__ __forceinline__ v8f vzero8() {
  v8f v;
#pragma unroll
  for (int i = 0; i < 8; ++i) v[i] = 0.0f;
  return v;
}

// A-fragment (16x32 bf16, row m per lane): K = kk*32 + half*8 + {0..7} and +16.
__device__ __forceinline__ v16bf ldfragA(const unsigned short* row, int kk, int half) {
  const int kb = kk * 32 + half * 8;
  FragU f;
  f.u[0] = *(const uint4*)(row + kb);
  f.u[1] = *(const uint4*)(row + kb + 16);
  return f.v;
}
// B-fragment (32x16 bf16, col n per lane): K = kk32 + half*8 + {0..7} and +16.
__device__ __forceinline__ v16bf ldfragB(const unsigned short* colrow, int kk32, int half) {
  const int kb = kk32 + half * 8;
  FragU f;
  f.u[0] = *(const uint4*)(colrow + kb);
  f.u[1] = *(const uint4*)(colrow + kb + 16);
  return f.v;
}

__device__ __forceinline__ v8f wmma_bf16(v16bf a, v16bf b, v8f c) {
  // args: (neg_a, A, neg_b, B, c_mod, C, reuse_a, reuse_b)
  return __builtin_amdgcn_wmma_f32_16x16x32_bf16(false, a, false, b, (short)0, c, false, false);
}

__device__ __forceinline__ float gelu_exact(float x) {
  return 0.5f * x * (1.0f + erff(x * 0.70710678118654752440f));
}

// ---------------- TDM helpers (CDNA5 Tensor Data Mover, ISA ch.8 D# layout) ----------------
#if USE_TDM
__device__ __forceinline__ int rfl(unsigned v) { return __builtin_amdgcn_readfirstlane((int)v); }

__device__ __forceinline__ void tdm_issue(u32x4 g0, i32x8 g1, i32x4 g2, i32x4 g3) {
#if __clang_major__ >= 23
  i32x8 gx;
#pragma unroll
  for (int i = 0; i < 8; ++i) gx[i] = 0;
  __builtin_amdgcn_tensor_load_to_lds(g0, g1, g2, g3, gx, 0);  // 6-arg (therock-10.0)
#else
  __builtin_amdgcn_tensor_load_to_lds(g0, g1, g2, g3, 0);      // 5-arg (ROCm 7.2)
#endif
}

// Contiguous 1-D tile of bf16 elements -> LDS.
__device__ __forceinline__ void tdm_load_1d(unsigned lds_byte_off, const void* gsrc, unsigned nelem) {
  unsigned long long ga = (unsigned long long)gsrc;
  u32x4 g0;
  g0[0] = (unsigned)rfl(1u);                                   // count=1, user descriptor
  g0[1] = (unsigned)rfl(lds_byte_off);                         // lds_addr (bytes)
  g0[2] = (unsigned)rfl((unsigned)ga);                         // global_addr[31:0]
  g0[3] = (unsigned)rfl(((unsigned)(ga >> 32) & 0x01FFFFFFu) | (2u << 30));  // addr hi | type=2
  i32x8 g1;
  g1[0] = rfl(1u << 16);                                       // data_size=2B (code 1)
  g1[1] = rfl((nelem & 0xFFFFu) << 16);                        // tensor_dim0 lo16
  g1[2] = rfl((nelem >> 16) | (1u << 16));                     // tensor_dim0 hi | tensor_dim1=1
  g1[3] = rfl((nelem & 0xFFFFu) << 16);                        // tile_dim0 = nelem
  g1[4] = rfl(1u);                                             // tile_dim1=1, tile_dim2=0
  g1[5] = rfl(nelem);                                          // tensor_dim0_stride lo
  g1[6] = rfl(0u);
  g1[7] = rfl(0u);
  i32x4 gz; gz[0] = gz[1] = gz[2] = gz[3] = 0;
  tdm_issue(g0, g1, gz, gz);
}

// Gather-mode: 16 rows of `rowlen` bf16 each, 16-bit row indices, -> LDS (packed).
__device__ __forceinline__ void tdm_gather16(unsigned lds_byte_off, const void* gbase,
                                             unsigned rowlen, unsigned nrows, const int* tok) {
  unsigned long long ga = (unsigned long long)gbase;
  u32x4 g0;
  g0[0] = (unsigned)rfl(1u | 0x80000000u);                     // count=1 | gather_mode (16-bit idx)
  g0[1] = (unsigned)rfl(lds_byte_off);
  g0[2] = (unsigned)rfl((unsigned)ga);
  g0[3] = (unsigned)rfl(((unsigned)(ga >> 32) & 0x01FFFFFFu) | (2u << 30));
  i32x8 g1;
  g1[0] = rfl(1u << 16);                                       // data_size=2B
  g1[1] = rfl((rowlen & 0xFFFFu) << 16);                       // tensor_dim0 lo16
  g1[2] = rfl((rowlen >> 16) | ((nrows & 0xFFFFu) << 16));     // tensor_dim0 hi | tensor_dim1 lo
  g1[3] = rfl((nrows >> 16) | (rowlen << 16));                 // tensor_dim1 hi | tile_dim0
  g1[4] = rfl(16u);                                            // tile_dim1 = #indices
  g1[5] = rfl(rowlen);                                         // tensor_dim0_stride
  g1[6] = rfl(0u);
  g1[7] = rfl(0u);
  i32x4 g2, g3;
#pragma unroll
  for (int i = 0; i < 4; ++i)
    g2[i] = rfl((unsigned)(tok[2 * i] & 0xFFFF) | ((unsigned)tok[2 * i + 1] << 16));
#pragma unroll
  for (int i = 0; i < 4; ++i)
    g3[i] = rfl((unsigned)(tok[8 + 2 * i] & 0xFFFF) | ((unsigned)tok[9 + 2 * i] << 16));
  tdm_issue(g0, g1, g2, g3);
}
#endif

// ---------------- prep kernels ----------------

__global__ void cvt_bf16(const float* __restrict__ src, unsigned short* __restrict__ dst, int n) {
  int i = blockIdx.x * blockDim.x + threadIdx.x;
  if (i < n) dst[i] = f2bf(src[i]);
}

// [R][C] fp32 -> [C][R] bf16 per matrix (blockIdx.z = matrix). R,C multiples of 32.
__global__ void transpose_cvt(const float* __restrict__ src, unsigned short* __restrict__ dst,
                              int R, int C) {
  __shared__ float tile[32][33];
  const size_t mat = (size_t)blockIdx.z * R * C;
  const int c0 = blockIdx.x * 32, r0 = blockIdx.y * 32;
  const int tx = threadIdx.x, ty = threadIdx.y;
#pragma unroll
  for (int i = 0; i < 32; i += 8)
    tile[ty + i][tx] = src[mat + (size_t)(r0 + ty + i) * C + (c0 + tx)];
  __syncthreads();
#pragma unroll
  for (int i = 0; i < 32; i += 8)
    dst[mat + (size_t)(c0 + ty + i) * R + (r0 + tx)] = f2bf(tile[tx][ty + i]);
}

__global__ void norm_experts(const float* __restrict__ lE, float* __restrict__ outE) {
  const int e = blockIdx.x;
  __shared__ float red[256];
  float ss = 0.0f;
  for (int d = threadIdx.x; d < kD; d += 256) { float v = lE[e * kD + d]; ss += v * v; }
  red[threadIdx.x] = ss;
  __syncthreads();
  for (int s = 128; s > 0; s >>= 1) {
    if ((int)threadIdx.x < s) red[threadIdx.x] += red[threadIdx.x + s];
    __syncthreads();
  }
  const float inv = 1.0f / fmaxf(sqrtf(red[0]), 1e-12f);
  for (int d = threadIdx.x; d < kD; d += 256) outE[e * kD + d] = lE[e * kD + d] * inv;
}

// ---------------- router: WMMA projection + fp32 softmax/top-2 ----------------

__global__ __launch_bounds__(256) void router_kernel(
    const unsigned short* __restrict__ xb, const unsigned short* __restrict__ wlb,
    const float* __restrict__ lnE, const float* __restrict__ bias,
    int* __restrict__ counts, int* __restrict__ tlist, float* __restrict__ plist) {
  __shared__ __align__(16) unsigned char smem[16 * kD * 4];  // phase1: bf16 A; phase2: f32 proj
  unsigned short* abuf = (unsigned short*)smem;
  float* projbuf = (float*)smem;

  const int tid = threadIdx.x;
  const int lane = tid & 31, wave = tid >> 5;
  const int half = lane >> 4, mn = lane & 15;
  const int t0 = blockIdx.x * 16;

#if USE_TDM
  if (wave == 0) {
    tdm_load_1d((unsigned)(unsigned long long)(const void*)abuf,
                xb + (size_t)t0 * kD, 16u * kD);
    __builtin_amdgcn_s_wait_tensorcnt(0);
  }
  __syncthreads();
#else
  {
    const uint4* s = (const uint4*)(xb + (size_t)t0 * kD);
    uint4* d = (uint4*)abuf;
    for (int i = tid; i < 16 * kD / 8; i += 256) d[i] = s[i];
  }
  __syncthreads();
#endif

  v8f acc[6];
#pragma unroll
  for (int j = 0; j < 6; ++j) acc[j] = vzero8();
  const unsigned short* arow = abuf + mn * kD;
  const unsigned short* bcols[6];
#pragma unroll
  for (int j = 0; j < 6; ++j) bcols[j] = wlb + (size_t)((wave * 6 + j) * 16 + mn) * kD;

  // software-pipelined K loop: fragments for kk+1 loaded before kk's WMMAs
  v16bf a = ldfragA(arow, 0, half);
  v16bf bf[6];
#pragma unroll
  for (int j = 0; j < 6; ++j) bf[j] = ldfragB(bcols[j], 0, half);
  for (int kk = 0; kk < kD / 32; ++kk) {
    const int kn = (kk + 1 < kD / 32) ? kk + 1 : kk;  // clamped (last iter reload is dead)
    v16bf anx = ldfragA(arow, kn, half);
    v16bf bnx[6];
#pragma unroll
    for (int j = 0; j < 6; ++j) bnx[j] = ldfragB(bcols[j], kn * 32, half);
#pragma unroll
    for (int j = 0; j < 6; ++j) acc[j] = wmma_bf16(a, bf[j], acc[j]);
    a = anx;
#pragma unroll
    for (int j = 0; j < 6; ++j) bf[j] = bnx[j];
  }
  __syncthreads();  // done reading abuf; reuse LDS as fp32 proj buffer
#pragma unroll
  for (int j = 0; j < 6; ++j) {
    int colb = (wave * 6 + j) * 16;
#pragma unroll
    for (int r = 0; r < 8; ++r) {
      int m = r + 8 * half;
      projbuf[m * kD + colb + mn] = acc[j][r];
    }
  }
  __syncthreads();

  // 16 lanes per token: l2 norm, cosine vs 8 normalized experts, softmax, top-2.
  const int token = tid >> 4, sub = tid & 15;
  float ss = 0.0f;
  for (int j = 0; j < kD / 16; ++j) {
    float v = projbuf[token * kD + sub + j * 16];
    ss += v * v;
  }
  ss += __shfl_xor(ss, 8, 32); ss += __shfl_xor(ss, 4, 32);
  ss += __shfl_xor(ss, 2, 32); ss += __shfl_xor(ss, 1, 32);
  const float inv = 1.0f / fmaxf(sqrtf(ss), 1e-12f);

  float score[kE];
#pragma unroll
  for (int e = 0; e < kE; ++e) {
    float dot = 0.0f;
    for (int j = 0; j < kD / 16; ++j) {
      int d = sub + j * 16;
      dot += projbuf[token * kD + d] * lnE[e * kD + d];
    }
    dot += __shfl_xor(dot, 8, 32); dot += __shfl_xor(dot, 4, 32);
    dot += __shfl_xor(dot, 2, 32); dot += __shfl_xor(dot, 1, 32);
    score[e] = dot * inv;
  }
  if (sub == 0) {
    const int t = t0 + token;
    float p[kE];
    float mx = -1e30f;
#pragma unroll
    for (int e = 0; e < kE; ++e) { p[e] = score[e] + bias[(size_t)t * kE + e]; mx = fmaxf(mx, p[e]); }
    float sum = 0.0f;
#pragma unroll
    for (int e = 0; e < kE; ++e) { p[e] = expf(p[e] - mx); sum += p[e]; }
    const float invs = 1.0f / sum;
    int i1 = 0; float p1 = -1.0f;
#pragma unroll
    for (int e = 0; e < kE; ++e) { float pv = p[e] * invs; if (pv > p1) { p1 = pv; i1 = e; } }
    int i2 = -1; float p2 = -1.0f;
#pragma unroll
    for (int e = 0; e < kE; ++e) { if (e == i1) continue; float pv = p[e] * invs; if (pv > p2) { p2 = pv; i2 = e; } }
    int pos1 = atomicAdd(&counts[i1], 1);
    tlist[i1 * kT + pos1] = (t << 1);     plist[i1 * kT + pos1] = p1;
    int pos2 = atomicAdd(&counts[i2], 1);
    tlist[i2 * kT + pos2] = (t << 1) | 1; plist[i2 * kT + pos2] = p2;
  }
}

// ---------------- sparse expert MLP: fused GEMM1 -> GELU -> GEMM2 ----------------

__global__ __launch_bounds__(256) void expert_mlp(
    const unsigned short* __restrict__ xb,
    const unsigned short* __restrict__ w1t, const unsigned short* __restrict__ w2t,
    const float* __restrict__ b1, const float* __restrict__ b2,
    const int* __restrict__ counts, const int* __restrict__ tlist, const float* __restrict__ plist,
    float* __restrict__ moe0, float* __restrict__ moe1) {
  const int e = blockIdx.y;
  const int n = counts[e];
  const int base = blockIdx.x * 16;
  if (base >= n) return;  // uniform early exit

  __shared__ __align__(16) unsigned short abuf[16 * kD];    // 24 KB gathered token rows (bf16)
  __shared__ __align__(16) unsigned short hbuf[16 * 256];   //  8 KB gelu(h) chunk (bf16)
  __shared__ int   stok[16];
  __shared__ int   srank[16];
  __shared__ float sp[16];
  __shared__ int   sval[16];

  const int tid = threadIdx.x, lane = tid & 31, wave = tid >> 5;
  const int half = lane >> 4, mn = lane & 15;

  if (tid < 16) {
    int idx = base + tid;
    int valid = (idx < n);
    int entry = tlist[e * kT + (valid ? idx : base)];
    stok[tid] = entry >> 1;
    srank[tid] = entry & 1;
    sp[tid] = valid ? plist[e * kT + idx] : 0.0f;
    sval[tid] = valid;
  }
  __syncthreads();

#if USE_TDM
  if (wave == 0) {  // TDM gather of 16 token rows straight into LDS
    tdm_gather16((unsigned)(unsigned long long)(const void*)abuf, xb, kD, kT, stok);
    __builtin_amdgcn_s_wait_tensorcnt(0);
  }
  __syncthreads();
#else
  for (int i = tid; i < 16 * (kD / 8); i += 256) {
    int r = i / (kD / 8), c = i - r * (kD / 8);
    ((uint4*)abuf)[i] = ((const uint4*)(xb + (size_t)stok[r] * kD))[c];
  }
  __syncthreads();
#endif

  const unsigned short* w1e = w1t + (size_t)e * kD * kH;  // [h][d] bf16
  const unsigned short* w2e = w2t + (size_t)e * kH * kD;  // [d][h] bf16

  v8f acc2[6];
#pragma unroll
  for (int j = 0; j < 6; ++j) acc2[j] = vzero8();

  const unsigned short* arow = abuf + mn * kD;
  const unsigned short* hrow = hbuf + mn * 256;
  const unsigned short* w2cols[6];
#pragma unroll
  for (int j = 0; j < 6; ++j) w2cols[j] = w2e + (size_t)((wave * 6 + j) * 16 + mn) * kH;

  for (int hc = 0; hc < kH / 256; ++hc) {
    // GEMM1: wave owns 2 column tiles; shared A fragment; pipelined K loop.
    const int col0 = hc * 256 + wave * 32 + mn;
    const unsigned short* b1c0 = w1e + (size_t)col0 * kD;
    const unsigned short* b1c1 = w1e + (size_t)(col0 + 16) * kD;
    __builtin_prefetch(b1c0 + (size_t)256 * kD, 0, 0);  // next chunk's columns
    __builtin_prefetch(b1c1 + (size_t)256 * kD, 0, 0);
    v8f acc1a = vzero8(), acc1b = vzero8();
    {
      v16bf a  = ldfragA(arow, 0, half);
      v16bf u0 = ldfragB(b1c0, 0, half);
      v16bf u1 = ldfragB(b1c1, 0, half);
      for (int kk = 0; kk < kD / 32; ++kk) {
        const int kn = (kk + 1 < kD / 32) ? kk + 1 : kk;
        v16bf anx = ldfragA(arow, kn, half);
        v16bf n0  = ldfragB(b1c0, kn * 32, half);
        v16bf n1  = ldfragB(b1c1, kn * 32, half);
        acc1a = wmma_bf16(a, u0, acc1a);
        acc1b = wmma_bf16(a, u1, acc1b);
        a = anx; u0 = n0; u1 = n1;
      }
    }
    const float bias1a = b1[e * kH + col0];
    const float bias1b = b1[e * kH + col0 + 16];
    const int cl0 = wave * 32 + mn;
#pragma unroll
    for (int r = 0; r < 8; ++r) {
      int m = r + 8 * half;
      hbuf[m * 256 + cl0]      = f2bf(gelu_exact(acc1a[r] + bias1a));
      hbuf[m * 256 + cl0 + 16] = f2bf(gelu_exact(acc1b[r] + bias1b));
    }
    __syncthreads();
    // GEMM2 partial: acc2 += h-chunk * W2[256-slice x 768]; pipelined K loop.
    {
      v16bf a = ldfragA(hrow, 0, half);
      v16bf bf[6];
#pragma unroll
      for (int j = 0; j < 6; ++j) bf[j] = ldfragB(w2cols[j], hc * 256, half);
      for (int kk = 0; kk < 256 / 32; ++kk) {
        const int kn = (kk + 1 < 256 / 32) ? kk + 1 : kk;
        v16bf anx = ldfragA(hrow, kn, half);
        v16bf bnx[6];
#pragma unroll
        for (int j = 0; j < 6; ++j) bnx[j] = ldfragB(w2cols[j], hc * 256 + kn * 32, half);
#pragma unroll
        for (int j = 0; j < 6; ++j) acc2[j] = wmma_bf16(a, bf[j], acc2[j]);
        a = anx;
#pragma unroll
        for (int j = 0; j < 6; ++j) bf[j] = bnx[j];
      }
    }
    __syncthreads();
  }

  // epilogue: (acc + b2) * gate -> rank slot buffer; padded rows masked out
#pragma unroll
  for (int j = 0; j < 6; ++j) {
    const int col = (wave * 6 + j) * 16 + mn;
    const float bias2 = b2[e * kD + col];
#pragma unroll
    for (int r = 0; r < 8; ++r) {
      int m = r + 8 * half;
      if (sval[m]) {
        float v = (acc2[j][r] + bias2) * sp[m];
        float* dst = srank[m] ? moe1 : moe0;
        dst[(size_t)stok[m] * kD + col] = v;
      }
    }
  }
}

// ---------------- shared expert + final combine ----------------

__global__ __launch_bounds__(256) void shared_combine(
    const unsigned short* __restrict__ xb,
    const unsigned short* __restrict__ sw1t, const unsigned short* __restrict__ sw2t,
    const float* __restrict__ sb1, const float* __restrict__ sb2,
    const float* __restrict__ moe0, const float* __restrict__ moe1,
    float* __restrict__ out) {
  const int t0 = blockIdx.x * 16;
  __shared__ __align__(16) unsigned short abuf[16 * kD];
  __shared__ __align__(16) unsigned short hbuf[16 * 256];
  const int tid = threadIdx.x, lane = tid & 31, wave = tid >> 5;
  const int half = lane >> 4, mn = lane & 15;

#if USE_TDM
  if (wave == 0) {
    tdm_load_1d((unsigned)(unsigned long long)(const void*)abuf,
                xb + (size_t)t0 * kD, 16u * kD);
    __builtin_amdgcn_s_wait_tensorcnt(0);
  }
  __syncthreads();
#else
  {
    const uint4* s = (const uint4*)(xb + (size_t)t0 * kD);
    for (int i = tid; i < 16 * kD / 8; i += 256) ((uint4*)abuf)[i] = s[i];
  }
  __syncthreads();
#endif

  v8f acc2[6];
#pragma unroll
  for (int j = 0; j < 6; ++j) acc2[j] = vzero8();
  const unsigned short* arow = abuf + mn * kD;
  const unsigned short* hrow = hbuf + mn * 256;
  const unsigned short* w2cols[6];
#pragma unroll
  for (int j = 0; j < 6; ++j) w2cols[j] = sw2t + (size_t)((wave * 6 + j) * 16 + mn) * kH;

  for (int hc = 0; hc < kH / 256; ++hc) {
    const int col0 = hc * 256 + wave * 32 + mn;
    const unsigned short* b1c0 = sw1t + (size_t)col0 * kD;
    const unsigned short* b1c1 = sw1t + (size_t)(col0 + 16) * kD;
    __builtin_prefetch(b1c0 + (size_t)256 * kD, 0, 0);
    __builtin_prefetch(b1c1 + (size_t)256 * kD, 0, 0);
    v8f acc1a = vzero8(), acc1b = vzero8();
    {
      v16bf a  = ldfragA(arow, 0, half);
      v16bf u0 = ldfragB(b1c0, 0, half);
      v16bf u1 = ldfragB(b1c1, 0, half);
      for (int kk = 0; kk < kD / 32; ++kk) {
        const int kn = (kk + 1 < kD / 32) ? kk + 1 : kk;
        v16bf anx = ldfragA(arow, kn, half);
        v16bf n0  = ldfragB(b1c0, kn * 32, half);
        v16bf n1  = ldfragB(b1c1, kn * 32, half);
        acc1a = wmma_bf16(a, u0, acc1a);
        acc1b = wmma_bf16(a, u1, acc1b);
        a = anx; u0 = n0; u1 = n1;
      }
    }
    const float bias1a = sb1[col0];
    const float bias1b = sb1[col0 + 16];
    const int cl0 = wave * 32 + mn;
#pragma unroll
    for (int r = 0; r < 8; ++r) {
      int m = r + 8 * half;
      hbuf[m * 256 + cl0]      = f2bf(gelu_exact(acc1a[r] + bias1a));
      hbuf[m * 256 + cl0 + 16] = f2bf(gelu_exact(acc1b[r] + bias1b));
    }
    __syncthreads();
    {
      v16bf a = ldfragA(hrow, 0, half);
      v16bf bf[6];
#pragma unroll
      for (int j = 0; j < 6; ++j) bf[j] = ldfragB(w2cols[j], hc * 256, half);
      for (int kk = 0; kk < 256 / 32; ++kk) {
        const int kn = (kk + 1 < 256 / 32) ? kk + 1 : kk;
        v16bf anx = ldfragA(hrow, kn, half);
        v16bf bnx[6];
#pragma unroll
        for (int j = 0; j < 6; ++j) bnx[j] = ldfragB(w2cols[j], hc * 256 + kn * 32, half);
#pragma unroll
        for (int j = 0; j < 6; ++j) acc2[j] = wmma_bf16(a, bf[j], acc2[j]);
        a = anx;
#pragma unroll
        for (int j = 0; j < 6; ++j) bf[j] = bnx[j];
      }
    }
    __syncthreads();
  }
#pragma unroll
  for (int j = 0; j < 6; ++j) {
    const int col = (wave * 6 + j) * 16 + mn;
    const float bias2 = sb2[col];
#pragma unroll
    for (int r = 0; r < 8; ++r) {
      int m = r + 8 * half;
      size_t idx = (size_t)(t0 + m) * kD + col;
      out[idx] = acc2[j][r] + bias2 + moe0[idx] + moe1[idx];
    }
  }
}

// ---------------- host ----------------

extern "C" void kernel_launch(void* const* d_in, const int* in_sizes, int n_in,
                              void* d_out, int out_size, void* d_ws, size_t ws_size,
                              hipStream_t stream) {
  (void)in_sizes; (void)n_in; (void)out_size; (void)ws_size;
  const float* x    = (const float*)d_in[0];
  const float* Wl   = (const float*)d_in[1];
  const float* lE   = (const float*)d_in[2];
  const float* bias = (const float*)d_in[3];
  const float* W1   = (const float*)d_in[4];
  const float* b1   = (const float*)d_in[5];
  const float* W2   = (const float*)d_in[6];
  const float* b2   = (const float*)d_in[7];
  const float* sW1  = (const float*)d_in[8];
  const float* sb1  = (const float*)d_in[9];
  const float* sW2  = (const float*)d_in[10];
  const float* sb2  = (const float*)d_in[11];
  float* out = (float*)d_out;

  char* ws = (char*)d_ws;
  size_t off = 0;
  auto take = [&](size_t bytes) -> char* {
    char* p = ws + off;
    off = (off + bytes + 255) & ~(size_t)255;
    return p;
  };
  unsigned short* xb   = (unsigned short*)take((size_t)kT * kD * 2);
  unsigned short* wlb  = (unsigned short*)take((size_t)kD * kD * 2);
  unsigned short* w1t  = (unsigned short*)take((size_t)kE * kD * kH * 2);
  unsigned short* w2t  = (unsigned short*)take((size_t)kE * kD * kH * 2);
  unsigned short* sw1t = (unsigned short*)take((size_t)kD * kH * 2);
  unsigned short* sw2t = (unsigned short*)take((size_t)kD * kH * 2);
  float* lnEbuf = (float*)take((size_t)kE * kD * 4);
  int*   counts = (int*)take(256);
  int*   tlist  = (int*)take((size_t)kE * kT * 4);
  float* plist  = (float*)take((size_t)kE * kT * 4);
  float* moe0   = (float*)take((size_t)kT * kD * 4);
  float* moe1   = (float*)take((size_t)kT * kD * 4);

  cvt_bf16<<<(kT * kD + 255) / 256, 256, 0, stream>>>(x, xb, kT * kD);
  cvt_bf16<<<(kD * kD + 255) / 256, 256, 0, stream>>>(Wl, wlb, kD * kD);
  dim3 tb(32, 8);
  transpose_cvt<<<dim3(kH / 32, kD / 32, kE), tb, 0, stream>>>(W1, w1t, kD, kH);
  transpose_cvt<<<dim3(kD / 32, kH / 32, kE), tb, 0, stream>>>(W2, w2t, kH, kD);
  transpose_cvt<<<dim3(kH / 32, kD / 32, 1), tb, 0, stream>>>(sW1, sw1t, kD, kH);
  transpose_cvt<<<dim3(kD / 32, kH / 32, 1), tb, 0, stream>>>(sW2, sw2t, kH, kD);
  norm_experts<<<kE, 256, 0, stream>>>(lE, lnEbuf);
  hipMemsetAsync(counts, 0, 256, stream);
  router_kernel<<<kT / 16, 256, 0, stream>>>(xb, wlb, lnEbuf, bias, counts, tlist, plist);
  expert_mlp<<<dim3(kT / 16, kE), 256, 0, stream>>>(xb, w1t, w2t, b1, b2, counts, tlist, plist, moe0, moe1);
  shared_combine<<<kT / 16, 256, 0, stream>>>(xb, sw1t, sw2t, sb1, sb2, moe0, moe1, out);
}
